// LSTMDecoder_84645215469615
// MI455X (gfx1250) — matrix operand order repaired and checked by reading, not compiled, not run on hardware
//
#include <hip/hip_runtime.h>
#include <hip/hip_bf16.h>

// Problem dims
#define VV 32000
#define EE 256
#define HH 512
#define BB 64
#define SS 49
#define TT 20
#define G4 2048   // 4*H

typedef _Float16 h8   __attribute__((ext_vector_type(8)));
typedef _Float16 v16h __attribute__((ext_vector_type(16)));
typedef float    v8f  __attribute__((ext_vector_type(8)));

// ---------------------------------------------------------------------------
// WMMA operand loaders per CDNA5 ISA §7.12.2 (16-bit A 16x32, B 32x16, C f32 16x16)
// A row-major [M x K]: lane m = m0 + (l&15); halves 0-7: k = k0+(l>>4)*8 .. +7
//                      halves 8-15: k = k0+16+(l>>4)*8 .. +7
static __device__ __forceinline__ v16h load_a16(const _Float16* A, int lda,
                                                int m0, int k0, int lane) {
  int m  = m0 + (lane & 15);
  int kb = k0 + ((lane >> 4) << 3);
  h8 lo = *(const h8*)(A + m * lda + kb);
  h8 hi = *(const h8*)(A + m * lda + kb + 16);
  v16h r;
#pragma unroll
  for (int i = 0; i < 8; ++i) { r[i] = lo[i]; r[i + 8] = hi[i]; }
  return r;
}
// B: weights row-major [N x K] (i.e. transposed-B GEMM): lane n = n0 + (l&15);
// halves 0-7: k = k0+(l>>4)*8 .. +7; halves 8-15: k = k0+16+(l>>4)*8 .. +7
static __device__ __forceinline__ v16h load_b16(const _Float16* W, int ldb,
                                                int n0, int k0, int lane) {
  int n  = n0 + (lane & 15);
  int kb = k0 + ((lane >> 4) << 3);
  h8 lo = *(const h8*)(W + n * ldb + kb);
  h8 hi = *(const h8*)(W + n * ldb + kb + 16);
  v16h r;
#pragma unroll
  for (int i = 0; i < 8; ++i) { r[i] = lo[i]; r[i + 8] = hi[i]; }
  return r;
}

static __device__ __forceinline__ v8f wmma_f16(v16h a, v16h b, v8f c) {
  return __builtin_amdgcn_wmma_f32_16x16x32_f16(false, a, false, b,
                                                (short)0, c, false, false);
}

static __device__ __forceinline__ float sigm(float x) { return 1.f / (1.f + expf(-x)); }

// ---------------------------------------------------------------------------
// Prep kernels
__global__ __launch_bounds__(256) void cvt_f16(_Float16* __restrict__ dst,
                                               const float* __restrict__ src, int n) {
  for (int i = blockIdx.x * blockDim.x + threadIdx.x; i < n; i += gridDim.x * blockDim.x)
    dst[i] = (_Float16)src[i];
}

// dst[n][0:ka] = A[n][:], dst[n][ka:ka+kb] = Bm[n][:]   (f16 pack of concatenated weights)
__global__ __launch_bounds__(256) void pack2_f16(_Float16* __restrict__ dst,
                                                 const float* __restrict__ A,
                                                 const float* __restrict__ Bm,
                                                 int ka, int kb, int nrows) {
  int K = ka + kb, total = nrows * K;
  for (int i = blockIdx.x * blockDim.x + threadIdx.x; i < total; i += gridDim.x * blockDim.x) {
    int n = i / K, k = i - n * K;
    float v = (k < ka) ? A[n * ka + k] : Bm[n * kb + (k - ka)];
    dst[i] = (_Float16)v;
  }
}

__global__ __launch_bounds__(256) void bias_sum(float* __restrict__ dst,
                                                const float* __restrict__ a,
                                                const float* __restrict__ b, int n) {
  int i = blockIdx.x * blockDim.x + threadIdx.x;
  if (i < n) dst[i] = a[i] + b[i];
}

// scoreImgB[bs][e] = attn_b[e] + dot(img[bs][:], attn_W[e][512:768])  (one-time, fp32)
__global__ __launch_bounds__(256) void score_img(const float* __restrict__ img,
                                                 const float* __restrict__ attn_W,
                                                 const float* __restrict__ attn_b,
                                                 float* __restrict__ sIB) {
  int idx = blockIdx.x * blockDim.x + threadIdx.x;
  if (idx >= BB * SS * EE) return;
  int e = idx & (EE - 1), bs = idx >> 8;
  const float4* ip = (const float4*)(img + bs * EE);
  const float4* wp = (const float4*)(attn_W + e * (HH + EE) + HH);
  float acc = attn_b[e];
#pragma unroll 8
  for (int k = 0; k < EE / 4; ++k) {
    float4 a = ip[k], w = wp[k];
    acc += a.x * w.x + a.y * w.y + a.z * w.z + a.w * w.w;
  }
  sIB[idx] = acc;
}

// ---------------------------------------------------------------------------
// Per-step attention + softmax + context + combine.  One workgroup per batch b.
__global__ __launch_bounds__(256) void attn_step(
    const float* __restrict__ img, const int* __restrict__ captions,
    const float* __restrict__ emb_W, const float* __restrict__ attn_W,
    const float* __restrict__ comb_W, const float* __restrict__ comb_b,
    const float* __restrict__ h1, const float* __restrict__ sIB,
    _Float16* __restrict__ xcur,   // [64][768]: write x into cols 0..255
    _Float16* __restrict__ x1,     // [64][1024]: write f16(h1) into cols 512..1023
    float* __restrict__ attn_out, int t) {
  __shared__ float sH[EE];
  __shared__ float sScore[SS];
  __shared__ float sCtx[EE];
  __shared__ float sWe[EE];
  int b = blockIdx.x, tid = threadIdx.x;

  // stage 1: scoreH[e] = dot(h1[b], attn_W[e][0:512])
  {
    const float4* hp = (const float4*)(h1 + b * HH);
    const float4* wp = (const float4*)(attn_W + tid * (HH + EE));
    float acc = 0.f;
#pragma unroll 8
    for (int k = 0; k < HH / 4; ++k) {
      float4 h = hp[k], w = wp[k];
      acc += h.x * w.x + h.y * w.y + h.z * w.z + h.w * w.w;
    }
    sH[tid] = acc;
  }
  // stage 1b: f16 copy of h1 into layer-1 GEMM operand (cols 512..1023)
  for (int j = tid; j < HH; j += 256)
    x1[b * 1024 + HH + j] = (_Float16)h1[b * HH + j];
  // stage 1c: embedding gather
  {
    int cap = captions[t * BB + b];
    sWe[tid] = emb_W[cap * EE + tid];
  }
  __syncthreads();

  // stage 2: scores[s] = sum_e tanh(scoreImgB + scoreH)
  for (int s = tid; s < SS; s += 256) {
    const float* p = sIB + (b * SS + s) * EE;
    float sc = 0.f;
    for (int e = 0; e < EE; ++e) sc += tanhf(p[e] + sH[e]);
    sScore[s] = sc;
  }
  __syncthreads();

  // stage 3: softmax over S=49 (tiny; single lane)
  if (tid == 0) {
    float mx = -1e30f;
    for (int s = 0; s < SS; ++s) mx = fmaxf(mx, sScore[s]);
    float sum = 0.f;
    for (int s = 0; s < SS; ++s) { float e = expf(sScore[s] - mx); sScore[s] = e; sum += e; }
    float inv = 1.f / sum;
    for (int s = 0; s < SS; ++s) sScore[s] *= inv;
  }
  __syncthreads();
  if (tid < SS) attn_out[(t * BB + b) * SS + tid] = sScore[tid];

  // stage 4: context[e] = sum_s aw[s] * img[b][s][e]
  {
    float acc = 0.f;
    for (int s = 0; s < SS; ++s) acc += sScore[s] * img[(b * SS + s) * EE + tid];
    sCtx[tid] = acc;
  }
  __syncthreads();

  // stage 5: x[n] = comb_b[n] + dot([we,ctx], comb_W[n][0:512])
  {
    const float* wr = comb_W + tid * (2 * EE);
    float acc = comb_b[tid];
    for (int k = 0; k < EE; ++k) acc += sWe[k] * wr[k];
    for (int k = 0; k < EE; ++k) acc += sCtx[k] * wr[EE + k];
    xcur[b * 768 + tid] = (_Float16)acc;
  }
}

// ---------------------------------------------------------------------------
// LSTM layer 0: gates = Xcat[64,768] @ W0c^T[2048,768] + b0 ; fused elementwise.
// grid = 16 WGs: (m-tile 0..3) x (j-chunk 0..3 of 128 hidden cols)
__global__ __launch_bounds__(256) void lstm_l0(
    const _Float16* __restrict__ xcur, _Float16* __restrict__ xnext,
    const _Float16* __restrict__ W0c, const float* __restrict__ b0s,
    float* __restrict__ c0, _Float16* __restrict__ x1) {
  __shared__ float g[16 * 512];   // 32 KB: 16 batches x (4 gates x 128 j)
  int mt = blockIdx.x & 3, jc = blockIdx.x >> 2;
  int m0 = mt * 16, j0 = jc * 128;
  int lane = threadIdx.x & 31, wave = threadIdx.x >> 5;

  v8f acc[4] = {};   // wave owns 4 of the 32 (gate,j) 16x16 tiles
  for (int k0 = 0; k0 < 768; k0 += 32) {
    v16h a = load_a16(xcur, 768, m0, k0, lane);
#pragma unroll
    for (int i = 0; i < 4; ++i) {
      int f = wave * 4 + i, q = f >> 3, tj = f & 7;
      int n0 = q * HH + j0 + tj * 16;
      v16h bm = load_b16(W0c, 768, n0, k0, lane);
      acc[i] = wmma_f16(a, bm, acc[i]);
    }
  }
#pragma unroll
  for (int i = 0; i < 4; ++i) {
    int f = wave * 4 + i, q = f >> 3, tj = f & 7;
    int n0 = q * HH + j0 + tj * 16;
    float bias = b0s[n0 + (lane & 15)];
    int col = q * 128 + tj * 16 + (lane & 15);
    int mr = (lane >> 4) << 3;
#pragma unroll
    for (int r = 0; r < 8; ++r) g[(mr + r) * 512 + col] = acc[i][r] + bias;
  }
  __syncthreads();

  int bl = threadIdx.x >> 4;            // 0..15 local batch
  int jb = (threadIdx.x & 15) * 8;      // 8 consecutive local j per thread
  int b = m0 + bl;
#pragma unroll
  for (int u = 0; u < 8; ++u) {
    int jj = jb + u, j = j0 + jj;
    float I = sigm(g[bl * 512 + 0 * 128 + jj]);
    float F = sigm(g[bl * 512 + 1 * 128 + jj]);
    float Gg = tanhf(g[bl * 512 + 2 * 128 + jj]);
    float O = sigm(g[bl * 512 + 3 * 128 + jj]);
    float cN = F * c0[b * HH + j] + I * Gg;
    float hN = O * tanhf(cN);
    c0[b * HH + j] = cN;
    x1[b * 1024 + j] = (_Float16)hN;           // layer-1 input (cols 0..511)
    xnext[b * 768 + EE + j] = (_Float16)hN;    // next-step layer-0 h operand
  }
}

// LSTM layer 1: gates = X1[64,1024] @ W1c^T[2048,1024] + b1 ; fused elementwise.
__global__ __launch_bounds__(256) void lstm_l1(
    const _Float16* __restrict__ x1, const _Float16* __restrict__ W1c,
    const float* __restrict__ b1s, float* __restrict__ c1,
    float* __restrict__ h1, float* __restrict__ hid_out, int t) {
  __shared__ float g[16 * 512];
  int mt = blockIdx.x & 3, jc = blockIdx.x >> 2;
  int m0 = mt * 16, j0 = jc * 128;
  int lane = threadIdx.x & 31, wave = threadIdx.x >> 5;

  v8f acc[4] = {};
  for (int k0 = 0; k0 < 1024; k0 += 32) {
    v16h a = load_a16(x1, 1024, m0, k0, lane);
#pragma unroll
    for (int i = 0; i < 4; ++i) {
      int f = wave * 4 + i, q = f >> 3, tj = f & 7;
      int n0 = q * HH + j0 + tj * 16;
      v16h bm = load_b16(W1c, 1024, n0, k0, lane);
      acc[i] = wmma_f16(a, bm, acc[i]);
    }
  }
#pragma unroll
  for (int i = 0; i < 4; ++i) {
    int f = wave * 4 + i, q = f >> 3, tj = f & 7;
    int n0 = q * HH + j0 + tj * 16;
    float bias = b1s[n0 + (lane & 15)];
    int col = q * 128 + tj * 16 + (lane & 15);
    int mr = (lane >> 4) << 3;
#pragma unroll
    for (int r = 0; r < 8; ++r) g[(mr + r) * 512 + col] = acc[i][r] + bias;
  }
  __syncthreads();

  int bl = threadIdx.x >> 4;
  int jb = (threadIdx.x & 15) * 8;
  int b = m0 + bl;
#pragma unroll
  for (int u = 0; u < 8; ++u) {
    int jj = jb + u, j = j0 + jj;
    float I = sigm(g[bl * 512 + 0 * 128 + jj]);
    float F = sigm(g[bl * 512 + 1 * 128 + jj]);
    float Gg = tanhf(g[bl * 512 + 2 * 128 + jj]);
    float O = sigm(g[bl * 512 + 3 * 128 + jj]);
    float cN = F * c1[b * HH + j] + I * Gg;
    float hN = O * tanhf(cN);
    c1[b * HH + j] = cN;
    h1[b * HH + j] = hN;
    hid_out[(t * BB + b) * HH + j] = hN;
  }
}

// p1 + ReLU -> Z f16 row (vocab-GEMM A operand). One WG per batch b.
__global__ __launch_bounds__(256) void p1_z(const float* __restrict__ h1,
                                            const float* __restrict__ p1W,
                                            const float* __restrict__ p1b,
                                            _Float16* __restrict__ Z, int t) {
  int b = blockIdx.x, e = threadIdx.x;
  const float4* hp = (const float4*)(h1 + b * HH);
  const float4* wp = (const float4*)(p1W + e * HH);
  float acc = p1b[e];
#pragma unroll 8
  for (int k = 0; k < HH / 4; ++k) {
    float4 h = hp[k], w = wp[k];
    acc += h.x * w.x + h.y * w.y + h.z * w.z + h.w * w.w;
  }
  Z[(t * BB + b) * EE + e] = (_Float16)fmaxf(acc, 0.f);
}

// ---------------------------------------------------------------------------
// Vocab projection: out[1280, 32000] = Z[1280,256] @ p2W^T + p2b  (WMMA f16)
// grid: x = 32000/256 = 125, y = 1280/64 = 20.
// Each wave: 2 n-tiles x 4 m-tiles = 8 WMMAs per 12 b128 loads (1.5 loads/WMMA;
// A slab reused across both n-tiles). Next k-step's B lines prefetched.
__global__ __launch_bounds__(256) void vocab_gemm(const _Float16* __restrict__ Z,
                                                  const _Float16* __restrict__ p2Wh,
                                                  const float* __restrict__ p2b,
                                                  float* __restrict__ out) {
  int lane = threadIdx.x & 31, wave = threadIdx.x >> 5;
  int n0 = blockIdx.x * 256 + wave * 32;   // wave owns n-tiles n0, n0+16
  int m_base = blockIdx.y * 64;
  v8f acc[2][4] = {};
  for (int k0 = 0; k0 < EE; k0 += 32) {
    if (k0 + 32 < EE) {   // pull next k-step's B lines toward the WGP cache
      __builtin_prefetch(p2Wh + (n0 + (lane & 15)) * EE + k0 + 32, 0, 1);
      __builtin_prefetch(p2Wh + (n0 + 16 + (lane & 15)) * EE + k0 + 32, 0, 1);
    }
    v16h a[4];
#pragma unroll
    for (int i = 0; i < 4; ++i)
      a[i] = load_a16(Z, EE, m_base + i * 16, k0, lane);
#pragma unroll
    for (int j = 0; j < 2; ++j) {
      v16h bm = load_b16(p2Wh, EE, n0 + j * 16, k0, lane);
#pragma unroll
      for (int i = 0; i < 4; ++i)
        acc[j][i] = wmma_f16(a[i], bm, acc[j][i]);
    }
  }
  int mr = (lane >> 4) << 3;
#pragma unroll
  for (int j = 0; j < 2; ++j) {
    int n = n0 + j * 16 + (lane & 15);
    float bias = p2b[n];
#pragma unroll
    for (int i = 0; i < 4; ++i)
#pragma unroll
      for (int r = 0; r < 8; ++r) {
        int m = m_base + i * 16 + mr + r;
        out[m * VV + n] = acc[j][i][r] + bias;
      }
  }
}

// ---------------------------------------------------------------------------
extern "C" void kernel_launch(void* const* d_in, const int* in_sizes, int n_in,
                              void* d_out, int out_size, void* d_ws, size_t ws_size,
                              hipStream_t stream) {
  const float* img     = (const float*)d_in[0];
  const int*   caps    = (const int*)d_in[1];
  const float* emb_W   = (const float*)d_in[2];
  const float* attn_W  = (const float*)d_in[3];
  const float* attn_b  = (const float*)d_in[4];
  const float* comb_W  = (const float*)d_in[5];
  const float* comb_b  = (const float*)d_in[6];
  const float* Wih0    = (const float*)d_in[7];
  const float* Whh0    = (const float*)d_in[8];
  const float* bih0    = (const float*)d_in[9];
  const float* bhh0    = (const float*)d_in[10];
  const float* Wih1    = (const float*)d_in[11];
  const float* Whh1    = (const float*)d_in[12];
  const float* bih1    = (const float*)d_in[13];
  const float* bhh1    = (const float*)d_in[14];
  const float* p1W     = (const float*)d_in[15];
  const float* p1b     = (const float*)d_in[16];
  const float* p2W     = (const float*)d_in[17];
  const float* p2b     = (const float*)d_in[18];

  float* outputs = (float*)d_out;                      // [T,B,V]
  float* hid_out = outputs + (size_t)TT * BB * VV;     // [T,B,H]
  float* att_out = hid_out + (size_t)TT * BB * HH;     // [T,B,S]

  // Workspace carve-out (256B aligned)
  char* ws = (char*)d_ws;
  auto carve = [&](size_t bytes) { char* p = ws; ws += (bytes + 255) & ~(size_t)255; return p; };
  _Float16* p2Wh = (_Float16*)carve((size_t)VV * EE * 2);    // 16.4 MB
  _Float16* W0c  = (_Float16*)carve((size_t)G4 * 768 * 2);   // 3.1 MB
  _Float16* W1c  = (_Float16*)carve((size_t)G4 * 1024 * 2);  // 4.2 MB
  float*    b0s  = (float*)carve(G4 * 4);
  float*    b1s  = (float*)carve(G4 * 4);
  float*    sIB  = (float*)carve((size_t)BB * SS * EE * 4);  // 3.2 MB
  float*    c0   = (float*)carve((size_t)BB * HH * 4);
  float*    c1   = (float*)carve((size_t)BB * HH * 4);
  float*    h1   = (float*)carve((size_t)BB * HH * 4);
  _Float16* xA   = (_Float16*)carve((size_t)BB * 768 * 2);
  _Float16* xB   = (_Float16*)carve((size_t)BB * 768 * 2);
  _Float16* x1   = (_Float16*)carve((size_t)BB * 1024 * 2);
  _Float16* Z    = (_Float16*)carve((size_t)TT * BB * EE * 2);

  // Zero recurrent state (poisoned workspace; must re-init every call)
  hipMemsetAsync(c0, 0, (size_t)BB * HH * 4, stream);
  hipMemsetAsync(c1, 0, (size_t)BB * HH * 4, stream);
  hipMemsetAsync(h1, 0, (size_t)BB * HH * 4, stream);
  hipMemsetAsync(xA, 0, (size_t)BB * 768 * 2, stream);
  hipMemsetAsync(xB, 0, (size_t)BB * 768 * 2, stream);
  hipMemsetAsync(x1, 0, (size_t)BB * 1024 * 2, stream);

  // Prep: f16 weight packs + bias sums + precomputed image attention scores
  cvt_f16<<<4096, 256, 0, stream>>>(p2Wh, p2W, VV * EE);
  pack2_f16<<<2048, 256, 0, stream>>>(W0c, Wih0, Whh0, 256, 512, G4);
  pack2_f16<<<2048, 256, 0, stream>>>(W1c, Wih1, Whh1, 512, 512, G4);
  bias_sum<<<(G4 + 255) / 256, 256, 0, stream>>>(b0s, bih0, bhh0, G4);
  bias_sum<<<(G4 + 255) / 256, 256, 0, stream>>>(b1s, bih1, bhh1, G4);
  score_img<<<(BB * SS * EE + 255) / 256, 256, 0, stream>>>(img, attn_W, attn_b, sIB);

  // Sequential recurrence
  for (int t = 0; t < TT; ++t) {
    _Float16* xcur  = (t & 1) ? xB : xA;
    _Float16* xnext = (t & 1) ? xA : xB;
    attn_step<<<BB, 256, 0, stream>>>(img, caps, emb_W, attn_W, comb_W, comb_b,
                                      h1, sIB, xcur, x1, att_out, t);
    lstm_l0<<<16, 256, 0, stream>>>(xcur, xnext, W0c, b0s, c0, x1);
    lstm_l1<<<16, 256, 0, stream>>>(x1, W1c, b1s, c1, h1, hid_out, t);
    p1_z<<<BB, 256, 0, stream>>>(h1, p1W, p1b, Z, t);
  }

  // Big parallel vocab GEMM (compute-dominant half of the model)
  vocab_gemm<<<dim3(VV / 256, (TT * BB) / 64), 256, 0, stream>>>(Z, p2Wh, p2b, outputs);
}
